// MultiHeadAttention_33784212750923
// MI455X (gfx1250) — matrix operand order repaired and checked by reading, not compiled, not run on hardware
//
#include <hip/hip_runtime.h>

// ---------------- types ----------------
typedef __bf16 bf16_t;
typedef __attribute__((ext_vector_type(16))) __bf16 v16bf;
typedef __attribute__((ext_vector_type(8)))  __bf16 v8bf;
typedef __attribute__((ext_vector_type(8)))  float  v8f;
typedef __attribute__((ext_vector_type(4)))  unsigned int v4u;
typedef __attribute__((ext_vector_type(8)))  int v8i;
typedef __attribute__((ext_vector_type(4)))  int v4i;

union V16 { v16bf v; v8bf h[2]; };

// ---------------- problem constants ----------------
constexpr int Bn = 4, Sn = 2048, Dm = 1024, Hn = 16, Dk = 64;
constexpr int Mrows = Bn * Sn;                // 8192 rows for projections

__device__ __forceinline__ unsigned short f2bfu(float f) {
  unsigned u = __float_as_uint(f);
  return (unsigned short)((u + 0x7FFFu + ((u >> 16) & 1u)) >> 16);   // RNE
}
__device__ __forceinline__ bf16_t f2bf(float f) {
  unsigned short h = f2bfu(f);
  return __builtin_bit_cast(bf16_t, h);
}

// Async copy: global memory -> LDS, 16B per lane, tracked by ASYNCcnt.
// GV mode (SADDR=NULL): LDS offset comes from the VDST VGPR, global address from the VADDR VGPR pair.
__device__ __forceinline__ void async_g2l_b128(unsigned lds_off, const void* gptr) {
  asm volatile("global_load_async_to_lds_b128 %0, %1, off"
               :: "v"(lds_off), "v"((unsigned long long)(uintptr_t)gptr)
               : "memory");
}
__device__ __forceinline__ void wait_async0() {
  asm volatile("s_wait_asynccnt 0" ::: "memory");
}

// ---------------- fp32 -> bf16 convert ----------------
__global__ __launch_bounds__(256)
void convert_f32_bf16(const float* __restrict__ in, bf16_t* __restrict__ out, int n4) {
  int i = blockIdx.x * 256 + threadIdx.x;
  if (i >= n4) return;
  float4 v = ((const float4*)in)[i];
  unsigned lo = (unsigned)f2bfu(v.x) | ((unsigned)f2bfu(v.y) << 16);
  unsigned hi = (unsigned)f2bfu(v.z) | ((unsigned)f2bfu(v.w) << 16);
  ((uint2*)out)[i] = make_uint2(lo, hi);
}

// ---------------- GEMM: Y[M,N] = X[M,K=1024] * W[N,K=1024]^T + bias ----------------
// Block tile 128x128, BK=32, 256 threads = 8 waves, each wave 32(M)x64(N).
// LDS rows padded to 80B (20 dwords, odd vs 64 banks) -> conflict-free b128 frag reads.
// Staging uses GLOBAL_LOAD_ASYNC_TO_LDS_B128 double-buffered, one barrier/iter.
constexpr int BK = 32;
constexpr int KSTEPS = Dm / BK;               // 32
constexpr int LDR = 80;                       // padded LDS row stride (64B data + 16B pad)
constexpr int BUFB = 2 * 128 * LDR;           // one double-buffer slab (A+B) = 20480B

template <bool SPLIT_HEADS>
__global__ __launch_bounds__(256)
void gemm_xwt(const bf16_t* __restrict__ X, const bf16_t* __restrict__ W,
              const float* __restrict__ bias, void* __restrict__ outp) {
  __shared__ unsigned char smem[2 * BUFB];    // [buf][A|B][128 rows][80B] = 40960B

  const int tid = threadIdx.x;
  const int Mbase = blockIdx.y * 128;
  const int Nbase = blockIdx.x * 128;
  const int lane = tid & 31, w = tid >> 5;
  const int wm = w >> 1, wn = w & 1;          // wave grid 4(M) x 2(N)
  const int lh = lane >> 4, ll = lane & 15;
  const unsigned lds0 = (unsigned)(uintptr_t)&smem[0];   // LDS aperture: low 32 bits = offset

  // issue this thread's 4 async 16B chunks (2 for A tile, 2 for B tile) of K-slab `ks`
  auto issue_async = [&](int ks, int buf) {
    const unsigned ab = lds0 + buf * BUFB;
    const unsigned bb = ab + 128 * LDR;
#pragma unroll
    for (int i = 0; i < 2; ++i) {
      int c = tid + i * 256, row = c >> 2, sub = c & 3;
      async_g2l_b128(ab + row * LDR + sub * 16,
                     X + (size_t)(Mbase + row) * Dm + ks * BK + sub * 8);
      async_g2l_b128(bb + row * LDR + sub * 16,
                     W + (size_t)(Nbase + row) * Dm + ks * BK + sub * 8);
    }
  };

  v8f zero = {0.f, 0.f, 0.f, 0.f, 0.f, 0.f, 0.f, 0.f};
  v8f acc[2][4];
#pragma unroll
  for (int mt = 0; mt < 2; ++mt)
#pragma unroll
    for (int nt = 0; nt < 4; ++nt) acc[mt][nt] = zero;

  issue_async(0, 0);
  wait_async0();
  __syncthreads();

  for (int ks = 0; ks < KSTEPS; ++ks) {
    unsigned char* Ab = smem + (ks & 1) * BUFB;
    unsigned char* Bb = Ab + 128 * LDR;
    const bool more = (ks + 1 < KSTEPS);
    if (more) issue_async(ks + 1, (ks + 1) & 1);  // async DMA next K-slab into other buffer

    V16 af[2], bfm[4];
#pragma unroll
    for (int mt = 0; mt < 2; ++mt) {          // A frag: row m, chunks K{kb..kb+7} & K{kb+16..kb+23}
      const unsigned char* p = Ab + (wm * 32 + mt * 16 + ll) * LDR + lh * 16;
      af[mt].h[0] = *(const v8bf*)(p);
      af[mt].h[1] = *(const v8bf*)(p + 32);
    }
#pragma unroll
    for (int nt = 0; nt < 4; ++nt) {          // B frag: col n = W row, chunks K{kb..kb+7} & K{kb+16..kb+23}
      const unsigned char* p = Bb + (wn * 64 + nt * 16 + ll) * LDR + lh * 16;
      bfm[nt].h[0] = *(const v8bf*)(p);
      bfm[nt].h[1] = *(const v8bf*)(p + 32);
    }
#pragma unroll
    for (int mt = 0; mt < 2; ++mt)
#pragma unroll
      for (int nt = 0; nt < 4; ++nt)
        acc[mt][nt] = __builtin_amdgcn_wmma_f32_16x16x32_bf16(
            false, af[mt].v, false, bfm[nt].v, (short)0, acc[mt][nt], false, false);

    if (more) {
      wait_async0();                           // own async chunks landed in LDS
      __syncthreads();                         // everyone's landed -> safe to consume
    }
  }

  // epilogue: bias add + store
  float bv[4];
#pragma unroll
  for (int nt = 0; nt < 4; ++nt) bv[nt] = bias[Nbase + wn * 64 + nt * 16 + ll];

#pragma unroll
  for (int mt = 0; mt < 2; ++mt)
#pragma unroll
    for (int nt = 0; nt < 4; ++nt)
#pragma unroll
      for (int r = 0; r < 8; ++r) {
        int row = Mbase + wm * 32 + mt * 16 + lh * 8 + r;     // = b*Sn + s
        int col = Nbase + wn * 64 + nt * 16 + ll;             // = h*Dk + d
        float val = acc[mt][nt][r] + bv[nt];
        if constexpr (SPLIT_HEADS) {
          int h = col >> 6, d = col & 63;
          int b = row >> 11, s = row & (Sn - 1);
          ((bf16_t*)outp)[(((size_t)(b * Hn + h) * Sn) + s) * Dk + d] = f2bf(val);
        } else {
          ((float*)outp)[(size_t)row * Dm + col] = val;
        }
      }
}

// ---------------- flash attention over head-split Q/K/V ----------------
// grid = (Sn/64, Bn*Hn); block = 128 (4 waves); wave owns 16 q-rows, full dv=64.
// K tile (32x64 bf16) is DMA'd by the Tensor Data Mover with LDS padding
// (pad 4 DWORDs every 32 DWORDs -> 144B row stride), V is staged transposed.
constexpr int TK = 32;
constexpr int KST = 144;   // K tile row stride (128B + 16 pad) == TDM pad config
constexpr int VST = 80;    // V^T row stride   (64B + 16 pad)
constexpr int PST = 80;    // P row stride     (64B + 16 pad)

__global__ __launch_bounds__(128)
void flash_attn(const bf16_t* __restrict__ Qp, const bf16_t* __restrict__ Kp,
                const bf16_t* __restrict__ Vp, const int* __restrict__ mask,
                bf16_t* __restrict__ ctx) {
  __shared__ unsigned char smem[32 * KST + 64 * VST + 4 * 16 * PST];
  unsigned char* Ksh = smem;
  unsigned char* Vsh = smem + 32 * KST;
  unsigned char* Psh = Vsh + 64 * VST;

  const int tid = threadIdx.x, lane = tid & 31, w = tid >> 5;
  const int lh = lane >> 4, ll = lane & 15;
  const int bh = blockIdx.y;
  const size_t base = (size_t)bh * Sn * Dk;
  const int qbase = blockIdx.x * 64 + w * 16;
  unsigned char* Pw = Psh + w * 16 * PST;     // wave-private P scratch

  // ---- TDM descriptor (invariant part). D# group1 fields (ISA 08 §8.4):
  //  data_size=1 (2B), pad_enable=1, pad_interval=4 (32 DW), pad_amount=3 (4 DW),
  //  tensor_dim0=64, tensor_dim1=32, tile_dim0=64, tile_dim1=32, dim0_stride=64.
  v8i g1;
  g1[0] = (1 << 16) | (1 << 20) | (4 << 22) | (3 << 25);
  g1[1] = (int)(64u << 16);   // [47:32]=atomic_barrier_addr(0), [63:48]=tensor_dim0 lo16
  g1[2] = (int)(32u << 16);   // [79:64]=tensor_dim0 hi16(0),    [95:80]... tensor_dim1 lo16=32
  g1[3] = (int)(64u << 16);   // [111:96]=tensor_dim1 hi16(0),   [127:112]=tile_dim0=64
  g1[4] = 32;                 // [143:128]=tile_dim1=32, [159:144]=tile_dim2=0
  g1[5] = 64;                 // [191:160]=tensor_dim0_stride lo32 = 64
  g1[6] = 0;                  // dim0_stride hi16, dim1_stride lo16
  g1[7] = 0;
  const v4i gz = {0, 0, 0, 0};
  const v8i gz8 = {0, 0, 0, 0, 0, 0, 0, 0};
  const unsigned kshOff = (unsigned)(uintptr_t)Ksh;        // LDS byte offset
  const unsigned long long gK0 = (unsigned long long)(uintptr_t)(Kp + base);

  // Q fragments live in registers for the whole pass (16 rows x 64 dk)
  V16 qf[2];
  const bf16_t* qrow = Qp + base + (size_t)(qbase + ll) * Dk;
#pragma unroll
  for (int ks = 0; ks < 2; ++ks) {
    qf[ks].h[0] = *(const v8bf*)(qrow + ks * 32 + lh * 8);
    qf[ks].h[1] = *(const v8bf*)(qrow + ks * 32 + lh * 8 + 16);
  }

  v8f zero = {0.f, 0.f, 0.f, 0.f, 0.f, 0.f, 0.f, 0.f};
  v8f o[4];
#pragma unroll
  for (int nt = 0; nt < 4; ++nt) o[nt] = zero;
  float mrow[8], lrow[8];
#pragma unroll
  for (int r = 0; r < 8; ++r) { mrow[r] = -1.0e30f; lrow[r] = 0.f; }

  const float SCL = 0.18033688011112042f;     // (1/sqrt(64)) * log2(e)

  for (int kt = 0; kt < Sn / TK; ++kt) {
    const int kb = kt * TK;
    __syncthreads();                           // previous tile fully consumed

    // K tile via Tensor Data Mover (one instruction, wave 0 only; EXEC ignored)
    if (w == 0) {
      unsigned long long ga = gK0 + (unsigned long long)kb * Dk * 2;
      v4u g0;
      g0[0] = 1u;                              // count=1, is_restore=0, gather off
      g0[1] = kshOff;                          // lds_addr
      g0[2] = (unsigned)ga;                    // global_addr[31:0]
      g0[3] = (unsigned)(ga >> 32) | (2u << 30); // global_addr[56:32] | type=2
      __builtin_amdgcn_tensor_load_to_lds(g0, g1, gz, gz, gz8, 0);
    }

    // stage V tile transposed (64 x 32 bf16) with all 128 threads
#pragma unroll
    for (int i = 0; i < 2; ++i) {
      int c = tid + i * 128, row = c >> 3, sub = c & 7;
      uint4 d4 = *(const uint4*)(Vp + base + (size_t)(kb + row) * Dk + sub * 8);
      const unsigned short* e = (const unsigned short*)&d4;
#pragma unroll
      for (int j = 0; j < 8; ++j)
        *(unsigned short*)(Vsh + (sub * 8 + j) * VST + row * 2) = e[j];
      if (kt + 1 < Sn / TK)                    // global_prefetch_b8 for next V tile
        __builtin_prefetch(Vp + base + (size_t)(kb + TK + row) * Dk + sub * 8, 0, 0);
    }

    if (w == 0) __builtin_amdgcn_s_wait_tensorcnt(0);  // K tile landed in LDS
    __syncthreads();

    // scores S = Q * K^T  (two 16x16 col-tiles, dk contraction in 2 steps)
    v8f sc[2];
    sc[0] = zero; sc[1] = zero;
#pragma unroll
    for (int nt = 0; nt < 2; ++nt)
#pragma unroll
      for (int ks = 0; ks < 2; ++ks) {
        V16 kf;
        const unsigned char* p = Ksh + (nt * 16 + ll) * KST + ks * 64 + lh * 16;
        kf.h[0] = *(const v8bf*)(p);
        kf.h[1] = *(const v8bf*)(p + 32);
        sc[nt] = __builtin_amdgcn_wmma_f32_16x16x32_bf16(
            false, qf[ks].v, false, kf.v, (short)0, sc[nt], false, false);
      }

    // scale + mask (exp2 domain)
    float x[2][8];
#pragma unroll
    for (int nt = 0; nt < 2; ++nt)
#pragma unroll
      for (int r = 0; r < 8; ++r) {
        int row = qbase + lh * 8 + r;
        int col = kb + nt * 16 + ll;
        int mk = mask[(size_t)row * Sn + col];
        float v = sc[nt][r] * SCL;
        x[nt][r] = mk ? v : -1.0e9f;
      }

    // online softmax: butterfly reductions within each 16-lane half (wave32)
    float alpha[8];
#pragma unroll
    for (int r = 0; r < 8; ++r) {
      float t = fmaxf(x[0][r], x[1][r]);
      t = fmaxf(t, __shfl_xor(t, 1, 32));
      t = fmaxf(t, __shfl_xor(t, 2, 32));
      t = fmaxf(t, __shfl_xor(t, 4, 32));
      t = fmaxf(t, __shfl_xor(t, 8, 32));
      float mnew = fmaxf(mrow[r], t);
      alpha[r] = exp2f(mrow[r] - mnew);
      mrow[r] = mnew;
      float p0 = exp2f(x[0][r] - mnew);
      float p1 = exp2f(x[1][r] - mnew);
      *(unsigned short*)(Pw + (lh * 8 + r) * PST + ll * 2) = f2bfu(p0);
      *(unsigned short*)(Pw + (lh * 8 + r) * PST + (16 + ll) * 2) = f2bfu(p1);
      float s = p0 + p1;
      s += __shfl_xor(s, 1, 32);
      s += __shfl_xor(s, 2, 32);
      s += __shfl_xor(s, 4, 32);
      s += __shfl_xor(s, 8, 32);
      lrow[r] = lrow[r] * alpha[r] + s;
    }
#pragma unroll
    for (int nt = 0; nt < 4; ++nt)
#pragma unroll
      for (int r = 0; r < 8; ++r) o[nt][r] *= alpha[r];

    // O += P * V  (P re-read as A-fragment from wave-private LDS; in-order DS pipe)
    V16 pf;
    const unsigned char* pp = Pw + ll * PST + lh * 16;
    pf.h[0] = *(const v8bf*)(pp);
    pf.h[1] = *(const v8bf*)(pp + 32);
#pragma unroll
    for (int nt = 0; nt < 4; ++nt) {
      V16 vf;
      const unsigned char* vp = Vsh + (nt * 16 + ll) * VST + lh * 16;
      vf.h[0] = *(const v8bf*)(vp);
      vf.h[1] = *(const v8bf*)(vp + 32);
      o[nt] = __builtin_amdgcn_wmma_f32_16x16x32_bf16(
          false, pf.v, false, vf.v, (short)0, o[nt], false, false);
    }
  }

  // epilogue: normalize, write ctx in [B,S,Dm] layout (bf16) for the O-projection
  const int b = bh >> 4, h = bh & 15;
#pragma unroll
  for (int r = 0; r < 8; ++r) {
    float inv = 1.0f / lrow[r];
    int qr = qbase + lh * 8 + r;
    bf16_t* orow = ctx + ((size_t)(b * Sn + qr)) * Dm + h * Dk;
#pragma unroll
    for (int nt = 0; nt < 4; ++nt) orow[nt * 16 + ll] = f2bf(o[nt][r] * inv);
  }
}

// ---------------- launcher ----------------
extern "C" void kernel_launch(void* const* d_in, const int* in_sizes, int n_in,
                              void* d_out, int out_size, void* d_ws, size_t ws_size,
                              hipStream_t stream) {
  const float* q  = (const float*)d_in[0];
  const float* k  = (const float*)d_in[1];
  const float* v  = (const float*)d_in[2];
  const int*   mk = (const int*)d_in[3];
  const float* Wq = (const float*)d_in[4];  const float* bq = (const float*)d_in[5];
  const float* Wk = (const float*)d_in[6];  const float* bk = (const float*)d_in[7];
  const float* Wv = (const float*)d_in[8];  const float* bv = (const float*)d_in[9];
  const float* Wo = (const float*)d_in[10]; const float* bo = (const float*)d_in[11];

  // workspace carve-up (all bf16 intermediates)
  size_t off = 0;
  auto carve = [&](size_t bytes) {
    void* p = (char*)d_ws + off;
    off += (bytes + 255) & ~(size_t)255;
    return p;
  };
  const size_t ACT = (size_t)Mrows * Dm * 2;  // 16 MiB
  const size_t WB  = (size_t)Dm * Dm * 2;     //  2 MiB
  bf16_t* qb  = (bf16_t*)carve(ACT);
  bf16_t* kb  = (bf16_t*)carve(ACT);
  bf16_t* vb  = (bf16_t*)carve(ACT);
  bf16_t* Wqb = (bf16_t*)carve(WB);
  bf16_t* Wkb = (bf16_t*)carve(WB);
  bf16_t* Wvb = (bf16_t*)carve(WB);
  bf16_t* Wob = (bf16_t*)carve(WB);
  bf16_t* Qp  = (bf16_t*)carve(ACT);          // [B,H,S,Dk]
  bf16_t* Kp  = (bf16_t*)carve(ACT);
  bf16_t* Vp  = (bf16_t*)carve(ACT);
  bf16_t* ctx = (bf16_t*)carve(ACT);          // [B,S,Dm]

  // 1) fp32 -> bf16
  {
    int n4a = Mrows * Dm / 4, n4w = Dm * Dm / 4;
    convert_f32_bf16<<<(n4a + 255) / 256, 256, 0, stream>>>(q, qb, n4a);
    convert_f32_bf16<<<(n4a + 255) / 256, 256, 0, stream>>>(k, kb, n4a);
    convert_f32_bf16<<<(n4a + 255) / 256, 256, 0, stream>>>(v, vb, n4a);
    convert_f32_bf16<<<(n4w + 255) / 256, 256, 0, stream>>>(Wq, Wqb, n4w);
    convert_f32_bf16<<<(n4w + 255) / 256, 256, 0, stream>>>(Wk, Wkb, n4w);
    convert_f32_bf16<<<(n4w + 255) / 256, 256, 0, stream>>>(Wv, Wvb, n4w);
    convert_f32_bf16<<<(n4w + 255) / 256, 256, 0, stream>>>(Wo, Wob, n4w);
  }

  // 2) projections (head-split bf16 outputs)
  dim3 ggrid(Dm / 128, Mrows / 128);          // (8, 64)
  gemm_xwt<true><<<ggrid, 256, 0, stream>>>(qb, Wqb, bq, (void*)Qp);
  gemm_xwt<true><<<ggrid, 256, 0, stream>>>(kb, Wkb, bk, (void*)Kp);
  gemm_xwt<true><<<ggrid, 256, 0, stream>>>(vb, Wvb, bv, (void*)Vp);

  // 3) attention
  dim3 agrid(Sn / 64, Bn * Hn);               // (32, 64)
  flash_attn<<<agrid, 128, 0, stream>>>(Qp, Kp, Vp, mk, ctx);

  // 4) output projection (fp32 out)
  gemm_xwt<false><<<ggrid, 256, 0, stream>>>(ctx, Wob, bo, d_out);
}
